// ElemwiseModels_240518169182
// MI455X (gfx1250) — matrix operand order repaired and checked by reading, not compiled, not run on hardware
//
#include <hip/hip_runtime.h>
#include <hip/hip_bf16.h>

#define N_ATOMS   131072
#define N_TYPES   4
#define DDESC     256
#define HDIM      256
#define ROWS_PER_BLOCK 128
#define PAD       128
#define PERM_CAP  (N_ATOMS + N_TYPES * PAD)
#define TILES_PER_TYPE 1024   // max ceil(131072/128)

typedef __attribute__((ext_vector_type(16))) __bf16 v16bf_t;
typedef __attribute__((ext_vector_type(8)))  float  v8f_t;

struct U32x8 { uint4 a, b; };

static __device__ __forceinline__ v16bf_t as_v16bf(uint4 a, uint4 b) {
    union { U32x8 u; v16bf_t v; } c;
    c.u.a = a; c.u.b = b;
    return c.v;
}

// Native f32 -> bf16 conversion (RTNE). On gfx1250 the plain cast lowers to the
// hardware cvt instead of the 4-op integer emulation.
static __device__ __forceinline__ unsigned short f2bf(float f) {
    union { __bf16 b; unsigned short u; } c;
    c.b = (__bf16)f;
    return c.u;
}

// Native gfx1250 V_TANH_F32 (single TRANS32 op, co-executes with WMMA).
static __device__ __forceinline__ float fast_tanh(float x) {
#if __has_builtin(__builtin_amdgcn_tanhf)
    return __builtin_amdgcn_tanhf(x);
#elif __has_builtin(__builtin_amdgcn_tanh_f32)
    return __builtin_amdgcn_tanh_f32(x);
#else
    return tanhf(x);
#endif
}

// ---------------------------------------------------------------- bucketing
// meta layout (ints): [0..3]=counts  [4..8]=paddedStart[5]  [9..12]=cursor
__global__ void k_init(int* meta, int* perm) {
    int i = blockIdx.x * blockDim.x + threadIdx.x;
    if (i < 16) meta[i] = 0;
    if (i < PERM_CAP) perm[i] = -1;
}

__global__ void k_count(const int* __restrict__ elems, int* meta) {
    int i = blockIdx.x * blockDim.x + threadIdx.x;
    if (i < N_ATOMS) atomicAdd(&meta[elems[i]], 1);
}

__global__ void k_scan(int* meta) {
    int off = 0;
    for (int t = 0; t < N_TYPES; ++t) {
        meta[4 + t] = off;          // padded segment start
        meta[9 + t] = off;          // scatter cursor
        off += (meta[t] + (PAD - 1)) & ~(PAD - 1);
    }
    meta[4 + N_TYPES] = off;
}

__global__ void k_scatter(const int* __restrict__ elems, int* meta, int* perm) {
    int i = blockIdx.x * blockDim.x + threadIdx.x;
    if (i < N_ATOMS) {
        int pos = atomicAdd(&meta[9 + elems[i]], 1);
        perm[pos] = i;
    }
}

// ------------------------------------------------- weight pack: fp32 -> bf16
// Packed B-fragment layout per (layer, type): element ((nt*8+ks)*32 + lane)*16 + h
// where lane = ((k%32)/16)*16 + (n%16), h = k%16   (B 32x16 KxN per WMMA step)
__global__ void k_pack(const float* __restrict__ W1, const float* __restrict__ W2,
                       unsigned short* __restrict__ packed) {
    int i = blockIdx.x * blockDim.x + threadIdx.x;   // 2*4*65536 threads
    int l = i >> 18;
    int r = i & 262143;
    int t = r >> 16;
    int e = r & 65535;
    int k = e >> 8;
    int n = e & 255;
    const float* W = l ? W2 : W1;
    float v = W[(t << 16) + (k << 8) + n];
    int nt   = n >> 4;
    int ks   = k >> 5;
    int rem  = k & 31;
    int lane = ((rem >> 4) << 4) | (n & 15);
    int h    = rem & 15;
    int idx  = ((l * N_TYPES + t) << 16) + (((nt * 8 + ks) * 32 + lane) << 4) + h;
    packed[idx] = f2bf(v);
}

// ------------------------------------------------------------------ the MLP
__global__ __launch_bounds__(256, 1)
void k_mlp(const float* __restrict__ desc,
           const int*   __restrict__ perm,
           const int*   __restrict__ meta,
           const unsigned short* __restrict__ packed,
           const float* __restrict__ b1,
           const float* __restrict__ b2,
           const float* __restrict__ W3,
           const float* __restrict__ b3,
           float* __restrict__ out)
{
    __shared__ unsigned short sA[ROWS_PER_BLOCK * DDESC];   // 64 KB, reused for h1

    const int t        = blockIdx.x / TILES_PER_TYPE;
    const int tile     = blockIdx.x % TILES_PER_TYPE;
    const int segStart = meta[4 + t];
    const int segLen   = meta[5 + t] - segStart;
    const int rowBase  = tile * ROWS_PER_BLOCK;
    if (rowBase >= segLen) return;                          // uniform exit

    const int wave = threadIdx.x >> 5;
    const int lane = threadIdx.x & 31;
    const int waveRow = rowBase + wave * 16;                // segment-relative

    // ---- stage 16 descriptor rows (this wave's M-tile) into LDS as bf16
    #pragma unroll
    for (int r = 0; r < 16; ++r) {
        int atom = perm[segStart + waveRow + r];
        int c = lane * 8;
        unsigned short* dst = &sA[(wave * 16 + r) * DDESC + c];
        uint4 o;
        if (atom >= 0) {
            const float4* p = (const float4*)(desc + (size_t)atom * DDESC + c);
            float4 a = p[0], b = p[1];
            o.x = (unsigned)f2bf(a.x) | ((unsigned)f2bf(a.y) << 16);
            o.y = (unsigned)f2bf(a.z) | ((unsigned)f2bf(a.w) << 16);
            o.z = (unsigned)f2bf(b.x) | ((unsigned)f2bf(b.y) << 16);
            o.w = (unsigned)f2bf(b.z) | ((unsigned)f2bf(b.w) << 16);
        } else {
            o.x = o.y = o.z = o.w = 0u;
        }
        *(uint4*)dst = o;
    }
    __syncthreads();

    const int relRow  = lane & 15;
    const int koff    = (lane >> 4) * 16;     // A-frag K sub-offset (contiguous K = 16h + s)
    const int colLane = lane & 15;
    const unsigned short* aRow = &sA[(wave * 16 + relRow) * DDESC];

    // ---- load all layer-1 A fragments (16x32 bf16 each) into registers
    v16bf_t afrag[8];
    #pragma unroll
    for (int ks = 0; ks < 8; ++ks) {
        const uint4* p = (const uint4*)(aRow + ks * 32 + koff);
        afrag[ks] = as_v16bf(p[0], p[1]);     // K:[off..off+16)
    }

    // ---- layer 1: h1 = tanh(x*W1 + b1), written back into sA as bf16
    const unsigned short* pw1 = packed + ((size_t)t << 16);
    #pragma unroll
    for (int nt = 0; nt < 16; ++nt) {
        int col = nt * 16 + colLane;
        v8f_t acc = {};                       // inline-0 C for first WMMA
        #pragma unroll
        for (int ks = 0; ks < 8; ++ks) {
            const uint4* bp = (const uint4*)(pw1 + (((nt * 8 + ks) * 32 + lane) << 4));
            v16bf_t bfrag = as_v16bf(bp[0], bp[1]);
            acc = __builtin_amdgcn_wmma_f32_16x16x32_bf16(
                false, afrag[ks], false, bfrag, (short)0, acc, false, false);
        }
        float bv = b1[t * HDIM + col];
        int rbase = (lane >> 4) * 8;
        #pragma unroll
        for (int v = 0; v < 8; ++v) {
            int row = rbase + v;
            sA[(wave * 16 + row) * DDESC + col] = f2bf(fast_tanh(acc[v] + bv));
        }
    }
    __syncthreads();

    // ---- layer 2 A fragments from h1
    v16bf_t a2[8];
    #pragma unroll
    for (int ks = 0; ks < 8; ++ks) {
        const uint4* p = (const uint4*)(aRow + ks * 32 + koff);
        a2[ks] = as_v16bf(p[0], p[1]);
    }

    // ---- layer 2 + layer 3 fused: partial[v] = sum_col tanh(h2)*W3[col]
    const unsigned short* pw2 = packed + (((size_t)N_TYPES + t) << 16);
    float partial[8];
    #pragma unroll
    for (int v = 0; v < 8; ++v) partial[v] = 0.f;

    #pragma unroll
    for (int nt = 0; nt < 16; ++nt) {
        int col = nt * 16 + colLane;
        v8f_t acc = {};
        #pragma unroll
        for (int ks = 0; ks < 8; ++ks) {
            const uint4* bp = (const uint4*)(pw2 + (((nt * 8 + ks) * 32 + lane) << 4));
            v16bf_t bfrag = as_v16bf(bp[0], bp[1]);
            acc = __builtin_amdgcn_wmma_f32_16x16x32_bf16(
                false, a2[ks], false, bfrag, (short)0, acc, false, false);
        }
        float bv  = b2[t * HDIM + col];
        float w3v = W3[t * HDIM + col];
        #pragma unroll
        for (int v = 0; v < 8; ++v)
            partial[v] += fast_tanh(acc[v] + bv) * w3v;
    }

    // ---- reduce over the 16 lanes of each half-wave (16 distinct cols each)
    #pragma unroll
    for (int v = 0; v < 8; ++v) {
        float s = partial[v];
        #pragma unroll
        for (int m = 1; m < 16; m <<= 1) s += __shfl_xor(s, m, 16);
        partial[v] = s;
    }

    if (colLane == 0) {
        int rbase = (lane >> 4) * 8;
        float bias = b3[t];
        #pragma unroll
        for (int v = 0; v < 8; ++v) {
            int atom = perm[segStart + waveRow + rbase + v];
            if (atom >= 0) out[atom] = partial[v] + bias;
        }
    }
}

// ----------------------------------------------------------------- launcher
extern "C" void kernel_launch(void* const* d_in, const int* in_sizes, int n_in,
                              void* d_out, int out_size, void* d_ws, size_t ws_size,
                              hipStream_t stream) {
    const float* desc  = (const float*)d_in[0];
    const int*   elems = (const int*)  d_in[1];
    const float* W1    = (const float*)d_in[2];
    const float* b1    = (const float*)d_in[3];
    const float* W2    = (const float*)d_in[4];
    const float* b2    = (const float*)d_in[5];
    const float* W3    = (const float*)d_in[6];
    const float* b3    = (const float*)d_in[7];
    float* out = (float*)d_out;

    char* ws = (char*)d_ws;
    int* meta = (int*)ws;                                   // 256 B reserved
    int* perm = (int*)(ws + 256);
    size_t permBytes = ((size_t)PERM_CAP * 4 + 255) & ~(size_t)255;
    unsigned short* packed = (unsigned short*)(ws + 256 + permBytes);  // 1 MB

    k_init   <<<(PERM_CAP + 255) / 256, 256, 0, stream>>>(meta, perm);
    k_count  <<<(N_ATOMS + 255) / 256, 256, 0, stream>>>(elems, meta);
    k_scan   <<<1, 1, 0, stream>>>(meta);
    k_scatter<<<(N_ATOMS + 255) / 256, 256, 0, stream>>>(elems, meta, perm);
    k_pack   <<<(2 * N_TYPES * 65536) / 256, 256, 0, stream>>>(W1, W2, packed);
    k_mlp    <<<N_TYPES * TILES_PER_TYPE, 256, 0, stream>>>(
        desc, perm, meta, packed, b1, b2, W3, b3, out);
}
